// OscillatorySSM_1460288881148
// MI455X (gfx1250) — matrix operand order repaired and checked by reading, not compiled, not run on hardware
//
#include <hip/hip_runtime.h>
#include <hip/hip_bf16.h>
#include <stdint.h>

// ---------------------------------------------------------------------------
// Model dims
// ---------------------------------------------------------------------------
#define LYR 8
#define BB  4
#define SS  2048
#define DD  256
#define SDX 512
#define TT  (BB * SS)          // 8192 tokens
#define K1  (2 * DD)           // 512  (xr||xi)
#define N1  (2 * SDX + SDX + 2 * DD)  // 2048 (Bx_r,Bx_i,gate,Dx_r,Dx_i)
#define K2  (2 * SDX)          // 1024 (hr||hi)
#define N2  (2 * DD)           // 512  (y_r,y_i)

typedef __attribute__((ext_vector_type(16))) __bf16 v16bf;
typedef __attribute__((ext_vector_type(8)))  __bf16 v8bf;
typedef __attribute__((ext_vector_type(8)))  float  v8f;
typedef __attribute__((ext_vector_type(4)))  unsigned int v4u;
typedef __attribute__((ext_vector_type(4)))  int    v4i;
typedef __attribute__((ext_vector_type(8)))  int    v8i;

union FragBf {
    v16bf v;
    v8bf  h[2];
};

#if defined(__has_builtin)
#if __has_builtin(__builtin_amdgcn_tensor_load_to_lds)
#define USE_TDM 1
#endif
#endif

// ---------------------------------------------------------------------------
// TDM 2D tile load: pack D# per CDNA5 ISA section 8 and issue tensor_load_to_lds.
// data_size = 2 (4-byte units). Tile = tile_cols_dw x tile_rows, row stride
// stride_dw, packed contiguously into LDS at lds_off.
// ---------------------------------------------------------------------------
#if defined(USE_TDM)
__device__ __forceinline__ void tdm_load_2d(uint32_t lds_off, const void* gptr,
                                            uint32_t tile_cols_dw, uint32_t tile_rows,
                                            uint32_t tensor_cols_dw, uint32_t tensor_rows,
                                            uint32_t stride_dw) {
    uint64_t ga = (uint64_t)(uintptr_t)gptr;
    v4u g0;
    g0[0] = 1u;                                   // count=1, user descriptor
    g0[1] = lds_off;                              // LDS byte address
    g0[2] = (uint32_t)ga;                         // global addr [31:0]
    g0[3] = (uint32_t)((ga >> 32) & 0x01FFFFFFu)  // global addr [56:32]
            | (2u << 30);                         // type = 2 ("image")
    v8i g1;
    g1[0] = (int)(2u << 16);                                          // data_size=4B, mask=0
    g1[1] = (int)((tensor_cols_dw & 0xFFFFu) << 16);                  // tensor_dim0[15:0]
    g1[2] = (int)(((tensor_cols_dw >> 16) & 0xFFFFu) |
                  ((tensor_rows & 0xFFFFu) << 16));                   // dim0 hi | dim1 lo
    g1[3] = (int)(((tensor_rows >> 16) & 0xFFFFu) |
                  ((tile_cols_dw & 0xFFFFu) << 16));                  // dim1 hi | tile_dim0
    g1[4] = (int)(tile_rows & 0xFFFFu);                               // tile_dim1 (tile_dim2=0)
    g1[5] = (int)stride_dw;                                           // tensor_dim0_stride lo
    g1[6] = 0;                                                        // stride hi | dim1_stride lo
    g1[7] = 0;
    v4i z4 = {0, 0, 0, 0};
#if __clang_major__ >= 23
    v8i z8 = {0, 0, 0, 0, 0, 0, 0, 0};
    __builtin_amdgcn_tensor_load_to_lds(g0, g1, z4, z4, z8, 0);
#else
    __builtin_amdgcn_tensor_load_to_lds(g0, g1, z4, z4, 0);
#endif
}
#endif

// ---------------------------------------------------------------------------
// WMMA GEMM: C[M,N](f32) = A[M,K](bf16, row-major) * BT[N,K](bf16, K-contig)
// Optional fused add of DX[m, dxOff + n].
// Block tile: 128 x (64*NT); 256 threads = 8 waves (2 M x 4 N);
// per-wave tile 64 x (16*NT): 4 x NT accumulators, 4+NT fragment loads / BK=32.
// ---------------------------------------------------------------------------
template <int NT>   // N 16-tiles per wave: 2 -> BN=128, 4 -> BN=256
__global__ __launch_bounds__(256)
__attribute__((amdgpu_waves_per_eu(2)))
void gemm_wmma_bf16(const __bf16* __restrict__ A, const __bf16* __restrict__ BT,
                    float* __restrict__ C, const float* __restrict__ DX,
                    int K, int ldc, int dxLd, int dxOff, int M, int N) {
    constexpr int BN    = NT * 64;
    constexpr int SLABN = NT * 16;
    __shared__ __align__(128) __bf16 sA[2][128 * 32];
    __shared__ __align__(128) __bf16 sB[2][BN * 32];

    const int tid  = threadIdx.x;
    const int wave = tid >> 5;
    const int lane = tid & 31;
    const int wm   = wave & 1;        // 0..1 -> 64-row slab
    const int wn   = wave >> 1;       // 0..3 -> SLABN-col slab
    const int blkM = blockIdx.y * 128;
    const int blkN = blockIdx.x * BN;
    const int KT   = K >> 5;          // BK = 32

    v8f acc[4][NT] = {};

#if defined(USE_TDM)
    const uint32_t ldsA0 = (uint32_t)(size_t)(&sA[0][0]);
    const uint32_t ldsA1 = (uint32_t)(size_t)(&sA[1][0]);
    const uint32_t ldsB0 = (uint32_t)(size_t)(&sB[0][0]);
    const uint32_t ldsB1 = (uint32_t)(size_t)(&sB[1][0]);
    const uint32_t kdw = (uint32_t)(K >> 1);     // row length in dwords
    if (tid == 0) {
        tdm_load_2d(ldsA0, A + (size_t)blkM * K, 16, 128, kdw, (uint32_t)M, kdw);
        tdm_load_2d(ldsB0, BT + (size_t)blkN * K, 16, BN, kdw, (uint32_t)N, kdw);
        if (KT > 1) {
            tdm_load_2d(ldsA1, A + (size_t)blkM * K + 32, 16, 128, kdw, (uint32_t)M, kdw);
            tdm_load_2d(ldsB1, BT + (size_t)blkN * K + 32, 16, BN, kdw, (uint32_t)N, kdw);
        }
    }
#endif

    for (int kt = 0; kt < KT; ++kt) {
        int cur;
#if defined(USE_TDM)
        cur = kt & 1;
        if (tid == 0) {
            if (kt + 1 < KT)
                __builtin_amdgcn_s_wait_tensorcnt(2);   // current buffer done
            else
                __builtin_amdgcn_s_wait_tensorcnt(0);
        }
        __syncthreads();
#else
        cur = 0;
        for (int cc = tid; cc < 512; cc += 256) {       // A: 128 rows x 64B
            int row = cc >> 2;
            int col = (cc & 3) * 8;
            *(uint4*)&sA[0][row * 32 + col] =
                *(const uint4*)(A + (size_t)(blkM + row) * K + kt * 32 + col);
        }
        for (int cc = tid; cc < BN * 4; cc += 256) {    // B: BN rows x 64B
            int row = cc >> 2;
            int col = (cc & 3) * 8;
            *(uint4*)&sB[0][row * 32 + col] =
                *(const uint4*)(BT + (size_t)(blkN + row) * K + kt * 32 + col);
        }
        __syncthreads();
#endif
        const __bf16* pA = &sA[cur][0];
        const __bf16* pB = &sB[cur][0];
        const int l15  = lane & 15;
        const int rsel = lane >> 4;

        // A fragment: lane<16 holds K {0..7,16..23}, lane>=16 holds K {8..15,24..31}
        FragBf a[4];
        const int kbA = rsel * 8;
#pragma unroll
        for (int mi = 0; mi < 4; ++mi) {
            int row = wm * 64 + mi * 16 + l15;
            a[mi].h[0] = *(const v8bf*)(pA + row * 32 + kbA);
            a[mi].h[1] = *(const v8bf*)(pA + row * 32 + 16 + kbA);
        }
        // B fragment: lane<16 holds K {0..7,16..23} of col n, lane>=16 holds K {8..15,24..31}
        FragBf b[NT];
        const int kbB = rsel * 8;
#pragma unroll
        for (int ni = 0; ni < NT; ++ni) {
            int col = wn * SLABN + ni * 16 + l15;
            b[ni].h[0] = *(const v8bf*)(pB + col * 32 + kbB);
            b[ni].h[1] = *(const v8bf*)(pB + col * 32 + 16 + kbB);
        }
#pragma unroll
        for (int mi = 0; mi < 4; ++mi)
#pragma unroll
            for (int ni = 0; ni < NT; ++ni)
                acc[mi][ni] = __builtin_amdgcn_wmma_f32_16x16x32_bf16(
                    false, a[mi].v, false, b[ni].v, (short)0, acc[mi][ni], false, false);

        __syncthreads();
#if defined(USE_TDM)
        if (tid == 0 && kt + 2 < KT) {
            const __bf16* ga = A + (size_t)blkM * K + (kt + 2) * 32;
            const __bf16* gb = BT + (size_t)blkN * K + (kt + 2) * 32;
            tdm_load_2d(cur ? ldsA1 : ldsA0, ga, 16, 128, kdw, (uint32_t)M, kdw);
            tdm_load_2d(cur ? ldsB1 : ldsB0, gb, 16, BN, kdw, (uint32_t)N, kdw);
        }
#endif
    }

    // Epilogue: C/D layout => lane holds column n, rows m0+{0..7} (+8 for lanes>=16)
    const int l15 = lane & 15;
    const int hi8 = (lane >> 4) * 8;
#pragma unroll
    for (int mi = 0; mi < 4; ++mi) {
#pragma unroll
        for (int ni = 0; ni < NT; ++ni) {
            int n = blkN + wn * SLABN + ni * 16 + l15;
            int mBase = blkM + wm * 64 + mi * 16 + hi8;
#pragma unroll
            for (int r = 0; r < 8; ++r) {
                int m = mBase + r;
                float v = acc[mi][ni][r];
                if (DX) v += DX[(size_t)m * dxLd + dxOff + n];
                C[(size_t)m * ldc + n] = v;
            }
        }
    }
}

// ---------------------------------------------------------------------------
// LayerNorm over D per component, pack [xr||xi] bf16 row of length 512.
// One block per token.
// ---------------------------------------------------------------------------
__global__ __launch_bounds__(256)
void ln_pack_bf16(const float* __restrict__ H, const float* __restrict__ g,
                  const float* __restrict__ b, __bf16* __restrict__ X) {
    const int t = blockIdx.x;
    const int d = threadIdx.x;
    float hr = H[(size_t)t * (2 * DD) + 2 * d];
    float hi = H[(size_t)t * (2 * DD) + 2 * d + 1];
    __shared__ float4 red[256];
    red[d] = make_float4(hr, hr * hr, hi, hi * hi);
    __syncthreads();
    for (int off = 128; off > 0; off >>= 1) {
        if (d < off) {
            float4 o = red[d + off], m = red[d];
            m.x += o.x; m.y += o.y; m.z += o.z; m.w += o.w;
            red[d] = m;
        }
        __syncthreads();
    }
    float4 s = red[0];
    const float inv = 1.0f / (float)DD;
    float mur = s.x * inv, mui = s.z * inv;
    float ir = rsqrtf(s.y * inv - mur * mur + 1e-5f);
    float ii = rsqrtf(s.w * inv - mui * mui + 1e-5f);
    float gg = g[d], bb = b[d];
    X[(size_t)t * K1 + d]      = (__bf16)((hr - mur) * ir * gg + bb);
    X[(size_t)t * K1 + DD + d] = (__bf16)((hi - mui) * ii * gg + bb);
}

// Final LayerNorm -> fp32 interleaved [B,S,D,2] output
__global__ __launch_bounds__(256)
void ln_final(const float* __restrict__ H, const float* __restrict__ g,
              const float* __restrict__ b, float* __restrict__ out) {
    const int t = blockIdx.x;
    const int d = threadIdx.x;
    float hr = H[(size_t)t * (2 * DD) + 2 * d];
    float hi = H[(size_t)t * (2 * DD) + 2 * d + 1];
    __shared__ float4 red[256];
    red[d] = make_float4(hr, hr * hr, hi, hi * hi);
    __syncthreads();
    for (int off = 128; off > 0; off >>= 1) {
        if (d < off) {
            float4 o = red[d + off], m = red[d];
            m.x += o.x; m.y += o.y; m.z += o.z; m.w += o.w;
            red[d] = m;
        }
        __syncthreads();
    }
    float4 s = red[0];
    const float inv = 1.0f / (float)DD;
    float mur = s.x * inv, mui = s.z * inv;
    float ir = rsqrtf(s.y * inv - mur * mur + 1e-5f);
    float ii = rsqrtf(s.w * inv - mui * mui + 1e-5f);
    float gg = g[d], bb = b[d];
    out[(size_t)t * (2 * DD) + 2 * d]     = (hr - mur) * ir * gg + bb;
    out[(size_t)t * (2 * DD) + 2 * d + 1] = (hi - mui) * ii * gg + bb;
}

// ---------------------------------------------------------------------------
// Build W1^T [N1=2048 rows, K1=512 cols] bf16 (K-contiguous):
// cols blocks: Bx_r | Bx_i | gate | Dx_r | Dx_i
// ---------------------------------------------------------------------------
__global__ __launch_bounds__(256)
void build_w1(const float* __restrict__ BWr, const float* __restrict__ BWi,
              const float* __restrict__ gW, const float* __restrict__ DWr,
              const float* __restrict__ DWi, __bf16* __restrict__ W) {
    int idx = blockIdx.x * 256 + threadIdx.x;     // N1*K1
    int n = idx >> 9;
    int k = idx & 511;
    float v;
    if (n < SDX) {
        v = (k < DD) ? BWr[k * SDX + n] : -BWi[(k - DD) * SDX + n];
    } else if (n < 2 * SDX) {
        int j = n - SDX;
        v = (k < DD) ? BWi[k * SDX + j] : BWr[(k - DD) * SDX + j];
    } else if (n < 3 * SDX) {
        int j = n - 2 * SDX;
        v = gW[k * SDX + j];
    } else if (n < 3 * SDX + DD) {
        int d = n - 3 * SDX;
        v = (k < DD) ? DWr[k * DD + d] : -DWi[(k - DD) * DD + d];
    } else {
        int d = n - 3 * SDX - DD;
        v = (k < DD) ? DWi[k * DD + d] : DWr[(k - DD) * DD + d];
    }
    W[(size_t)n * K1 + k] = (__bf16)v;
}

// Build W2^T [N2=512 rows, K2=1024 cols] bf16: cols blocks y_r | y_i
__global__ __launch_bounds__(256)
void build_w2(const float* __restrict__ CWr, const float* __restrict__ CWi,
              __bf16* __restrict__ W) {
    int idx = blockIdx.x * 256 + threadIdx.x;     // N2*K2
    int n = idx >> 10;
    int k = idx & 1023;
    float v;
    if (n < DD) {
        v = (k < SDX) ? CWr[k * DD + n] : -CWi[(k - SDX) * DD + n];
    } else {
        int d = n - DD;
        v = (k < SDX) ? CWi[k * DD + d] : CWr[(k - SDX) * DD + d];
    }
    W[(size_t)n * K2 + k] = (__bf16)v;
}

// ---------------------------------------------------------------------------
// Sequential scan: one thread per (b, sd) channel; 2048 steps.
// Reads Bx_r/Bx_i/gate from G1 rows, writes h_all as bf16 [hr||hi] + h_fin.
// ---------------------------------------------------------------------------
__global__ __launch_bounds__(256)
void scan_kernel(const float* __restrict__ G1, const float* __restrict__ h0,
                 const float* __restrict__ theta, const float* __restrict__ damp_p,
                 const float* __restrict__ gate_b, __bf16* __restrict__ Hbf,
                 float* __restrict__ hid_out) {
    int c = blockIdx.x * 256 + threadIdx.x;       // B*SD = 2048
    int b = c >> 9;
    int j = c & 511;
    float th = theta[j];
    float ct = __cosf(th), st = __sinf(th);
    float damp = 0.5f + 0.5f / (1.0f + __expf(-damp_p[j]));
    float gb = gate_b[j];
    float hr = h0[((size_t)b * SDX + j) * 2];
    float hi = h0[((size_t)b * SDX + j) * 2 + 1];
    const size_t base = (size_t)b * SS * N1;
#pragma unroll 4
    for (int s = 0; s < SS; ++s) {
        const float* row = G1 + base + (size_t)s * N1;
        float bxr = row[j];
        float bxi = row[SDX + j];
        float gp  = row[2 * SDX + j];
        float g = 1.0f / (1.0f + __expf(-(gp + gb)));
        float rr = (hr * ct - hi * st) * damp;
        float ri = (hr * st + hi * ct) * damp;
        hr = g * bxr + (1.0f - g) * rr;
        hi = g * bxi + (1.0f - g) * ri;
        size_t t = (size_t)b * SS + s;
        Hbf[t * K2 + j]       = (__bf16)hr;
        Hbf[t * K2 + SDX + j] = (__bf16)hi;
    }
    hid_out[((size_t)b * SDX + j) * 2]     = hr;
    hid_out[((size_t)b * SDX + j) * 2 + 1] = hi;
}

// out = y_r*y + y ; H += 0.1*out  (residual stream update)
__global__ __launch_bounds__(256)
void fuse_out(const float* __restrict__ Y, float* __restrict__ H) {
    int i = blockIdx.x * 256 + threadIdx.x;       // T*D
    int t = i >> 8;
    int d = i & 255;
    float yr = Y[(size_t)t * N2 + d];
    float yi = Y[(size_t)t * N2 + DD + d];
    size_t base = (size_t)t * (2 * DD) + 2 * d;
    H[base]     += 0.1f * (yr * yr + yr);
    H[base + 1] += 0.1f * (yr * yi + yi);
}

__global__ __launch_bounds__(256)
void copy_f32(const float* __restrict__ src, float* __restrict__ dst, int n) {
    int i = blockIdx.x * 256 + threadIdx.x;
    if (i < n) dst[i] = src[i];
}

// ---------------------------------------------------------------------------
extern "C" void kernel_launch(void* const* d_in, const int* in_sizes, int n_in,
                              void* d_out, int out_size, void* d_ws, size_t ws_size,
                              hipStream_t stream) {
    (void)in_sizes; (void)n_in; (void)out_size; (void)ws_size;
    const float* x      = (const float*)d_in[0];
    const float* h0     = (const float*)d_in[1];
    const float* theta  = (const float*)d_in[2];
    const float* damp_p = (const float*)d_in[3];
    const float* B_Wr   = (const float*)d_in[4];
    const float* B_Wi   = (const float*)d_in[5];
    const float* C_Wr   = (const float*)d_in[6];
    const float* C_Wi   = (const float*)d_in[7];
    const float* D_Wr   = (const float*)d_in[8];
    const float* D_Wi   = (const float*)d_in[9];
    const float* gate_W = (const float*)d_in[10];
    const float* gate_b = (const float*)d_in[11];
    const float* norm_g = (const float*)d_in[12];
    const float* norm_b = (const float*)d_in[13];
    const float* out_g  = (const float*)d_in[14];
    const float* out_b  = (const float*)d_in[15];

    char* ws = (char*)d_ws;
    size_t off = 0;
    float*  Hcur = (float*)(ws + off);  off += (size_t)TT * 2 * DD * 4;   // 16 MB
    __bf16* Xbf  = (__bf16*)(ws + off); off += (size_t)TT * K1 * 2;       // 8 MB
    __bf16* W1T  = (__bf16*)(ws + off); off += (size_t)N1 * K1 * 2;       // 2 MB
    __bf16* W2T  = (__bf16*)(ws + off); off += (size_t)N2 * K2 * 2;       // 1 MB
    float*  G1   = (float*)(ws + off);  off += (size_t)TT * N1 * 4;       // 64 MB
    __bf16* Hbf  = (__bf16*)(ws + off); off += (size_t)TT * K2 * 2;       // 16 MB
    float*  Yws  = (float*)(ws + off);  off += (size_t)TT * N2 * 4;       // 16 MB

    float* out_main = (float*)d_out;
    float* out_hid  = out_main + (size_t)TT * 2 * DD;

    // residual stream <- x
    {
        int n = TT * 2 * DD;
        copy_f32<<<(n + 255) / 256, 256, 0, stream>>>(x, Hcur, n);
    }

    for (int i = 0; i < LYR; ++i) {
        ln_pack_bf16<<<TT, 256, 0, stream>>>(Hcur, norm_g + i * DD, norm_b + i * DD, Xbf);

        build_w1<<<(N1 * K1) / 256, 256, 0, stream>>>(
            B_Wr + (size_t)i * DD * SDX, B_Wi + (size_t)i * DD * SDX,
            gate_W + (size_t)i * 2 * DD * SDX,
            D_Wr + (size_t)i * DD * DD, D_Wi + (size_t)i * DD * DD, W1T);

        // 128x256 blocks: 8 x 64 = 512 workgroups, 16 WMMA / K-step / wave
        gemm_wmma_bf16<4><<<dim3(N1 / 256, TT / 128), 256, 0, stream>>>(
            Xbf, W1T, G1, nullptr, K1, N1, 0, 0, TT, N1);

        scan_kernel<<<(BB * SDX) / 256, 256, 0, stream>>>(
            G1, h0 + (size_t)i * BB * SDX * 2, theta + (size_t)i * SDX,
            damp_p + (size_t)i * SDX, gate_b + (size_t)i * SDX,
            Hbf, out_hid + (size_t)i * BB * SDX * 2);

        build_w2<<<(N2 * K2) / 256, 256, 0, stream>>>(
            C_Wr + (size_t)i * SDX * DD, C_Wi + (size_t)i * SDX * DD, W2T);

        // y = h_all @ Wc + Dx  (Dx lives in G1 columns [1536, 2048); col = 1536+n)
        // 128x128 blocks: 4 x 64 = 256 workgroups for WGP coverage
        gemm_wmma_bf16<2><<<dim3(N2 / 128, TT / 128), 256, 0, stream>>>(
            Hbf, W2T, Yws, G1, K2, N2, N1, 3 * SDX, TT, N2);

        fuse_out<<<(TT * DD) / 256, 256, 0, stream>>>(Yws, Hcur);
    }

    ln_final<<<TT, 256, 0, stream>>>(Hcur, out_g, out_b, out_main);
}
